// SingleLayerMMoE_87265145520754
// MI455X (gfx1250) — compile-verified, matrix operand order checked by reading, not run on hardware
//
#include <hip/hip_runtime.h>

// ---------------------------------------------------------------------------
// MMoE single layer on gfx1250 (CDNA5, wave32, WMMA, async LDS staging).
//
// Pipeline:
//   k_prep      : pack W1/W2/towerW1 -> bf16 (transposed for LDS [N][K]), zero stats
//   k_cvt_x     : x fp32 -> bf16
//   k_gemm1     : h1 = x @ W1cat  (v_wmma_f32_16x16x32_bf16), bf16 out + BN1 sums
//                 A/B tiles staged with GLOBAL_LOAD_ASYNC_TO_LDS_B128 (ASYNCcnt)
//   k_finalize  : BN stats -> per-channel scale/shift (folds gamma/beta)
//   k_gemm2     : per-expert h2 = silu(bn1(h1_e)) @ W2_e, bf16 out + BN2 sums
//                 B tile async; A tile register path (fused BN+SiLU) + prefetch
//   k_finalize  : BN2 scale/shift
//   k_epilogue  : gate softmax + bn2/silu + interest mix + tower MLP -> out[B]
// ---------------------------------------------------------------------------

typedef __attribute__((ext_vector_type(16))) __bf16 v16bf;
typedef __attribute__((ext_vector_type(8)))  __bf16 v8bf;
typedef __attribute__((ext_vector_type(8)))  float  v8f;

#define B_N   65536
#define EMB   512
#define H_DIM 256
#define O_DIM 256
#define D_N   3
#define E_N   4
#define TH_N  128
#define EH    (E_N * H_DIM)   // 1024
#define EO    (E_N * O_DIM)   // 1024

#define BM  128
#define BN  128
#define BK  32
#define LDT 40                // padded LDS row stride in bf16 (80B, 16B-aligned chunks)

// --- CDNA5 async LDS staging helpers ---------------------------------------
// LDS operand = allocation-relative offset = low 32 bits of the flat shared
// address (ISA: shared aperture occupies addr[63:32] only).
__device__ __forceinline__ unsigned lds_off(const void* p) {
  return (unsigned)(uintptr_t)p;
}
// GVS mode: mem_addr = SADDR(64) + VADDR(32, per-lane byte offset)
__device__ __forceinline__ void async_copy_b128(unsigned lds, unsigned goff,
                                                const void* base) {
  asm volatile("global_load_async_to_lds_b128 %0, %1, %2"
               :: "v"(lds), "v"(goff), "s"(base)
               : "memory");
}
__device__ __forceinline__ void wait_async0() {
  asm volatile("s_wait_asynccnt 0x0" ::: "memory");
}

// ---------------------------------------------------------------------------
__global__ void k_prep(const float* __restrict__ W1, const float* __restrict__ W2,
                       const float* __restrict__ TW1,
                       __bf16* __restrict__ w1t, __bf16* __restrict__ w2t,
                       __bf16* __restrict__ tw1t, float* __restrict__ stats) {
  int nt = gridDim.x * blockDim.x;
  int t  = blockIdx.x * blockDim.x + threadIdx.x;
  // w1t[n][k] = W1[e, k, h]  with n = e*H + h   (N-major, K contiguous)
  for (int i = t; i < EH * EMB; i += nt) {
    int n = i / EMB, k = i - n * EMB;
    int e = n / H_DIM, h = n - e * H_DIM;
    w1t[i] = (__bf16)W1[(size_t)e * EMB * H_DIM + (size_t)k * H_DIM + h];
  }
  // w2t[e][o][h] = W2[e, h, o]
  for (int i = t; i < E_N * O_DIM * H_DIM; i += nt) {
    int e = i / (O_DIM * H_DIM); int r = i - e * O_DIM * H_DIM;
    int o = r / H_DIM, h = r - o * H_DIM;
    w2t[i] = (__bf16)W2[(size_t)e * H_DIM * O_DIM + (size_t)h * O_DIM + o];
  }
  // tw1t[d][th][o] = TW1[d, o, th]
  for (int i = t; i < D_N * TH_N * O_DIM; i += nt) {
    int d = i / (TH_N * O_DIM); int r = i - d * TH_N * O_DIM;
    int th = r / O_DIM, o = r - th * O_DIM;
    tw1t[i] = (__bf16)TW1[(size_t)d * O_DIM * TH_N + (size_t)o * TH_N + th];
  }
  for (int i = t; i < 8 * 1024; i += nt) stats[i] = 0.0f;
}

__global__ void k_cvt_x(const float* __restrict__ x, __bf16* __restrict__ xb) {
  size_t t  = (size_t)blockIdx.x * blockDim.x + threadIdx.x;
  size_t nt = (size_t)gridDim.x * blockDim.x;
  for (size_t i = t; i < (size_t)B_N * EMB; i += nt) xb[i] = (__bf16)x[i];
}

// load a 16x32 bf16 WMMA A/B fragment from LDS tile stored [rows][LDT]
__device__ __forceinline__ v16bf load_frag(const __bf16* base, int lane) {
  int r  = lane & 15;
  int kh = (lane >> 4) << 3;              // 0 or 8
  union { v16bf v; v8bf h[2]; } u;
  u.h[0] = *(const v8bf*)(base + r * LDT + kh);       // K kh..kh+7
  u.h[1] = *(const v8bf*)(base + r * LDT + kh + 16);  // K kh+16..kh+23
  return u.v;
}

// ---------------------------------------------------------------------------
__global__ __launch_bounds__(256) void k_gemm1(
    const __bf16* __restrict__ xb, const __bf16* __restrict__ w1t,
    __bf16* __restrict__ h1, float* __restrict__ sum1, float* __restrict__ sq1) {
  __shared__ __attribute__((aligned(16))) __bf16 sA[BM * LDT];
  __shared__ __attribute__((aligned(16))) __bf16 sB[BN * LDT];
  const int rowBase = blockIdx.x * BM;
  const int colBase = blockIdx.y * BN;
  const int lane = threadIdx.x & 31;
  const int wave = threadIdx.x >> 5;
  const int wm = wave >> 1;               // 4 wave rows  -> 32 M each
  const int wn = wave & 1;                // 2 wave cols  -> 64 N each
  v8f acc[2][4] = {};

  for (int kk = 0; kk < EMB; kk += BK) {
#pragma unroll
    for (int i = 0; i < 2; i++) {         // 256 thr * 2 * 8 bf16 = 128x32 tile
      int c = threadIdx.x + 256 * i;
      int r = c >> 2;
      int k8 = (c & 3) << 3;
      unsigned la = lds_off(&sA[r * LDT + k8]);
      unsigned lb = lds_off(&sB[r * LDT + k8]);
      unsigned ga = (unsigned)(((rowBase + r) * EMB + kk + k8) * 2);
      unsigned gb = (unsigned)(((colBase + r) * EMB + kk + k8) * 2);
      async_copy_b128(la, ga, xb);        // memory -> LDS, no VGPR round trip
      async_copy_b128(lb, gb, w1t);
    }
    wait_async0();
    __syncthreads();
    v16bf af[2], bf[4];
#pragma unroll
    for (int m = 0; m < 2; m++) af[m] = load_frag(&sA[(wm * 32 + m * 16) * LDT], lane);
#pragma unroll
    for (int n = 0; n < 4; n++) bf[n] = load_frag(&sB[(wn * 64 + n * 16) * LDT], lane);
#pragma unroll
    for (int m = 0; m < 2; m++)
#pragma unroll
      for (int n = 0; n < 4; n++)
        acc[m][n] = __builtin_amdgcn_wmma_f32_16x16x32_bf16(
            false, af[m], false, bf[n], (short)0, acc[m][n], false, false);
    __syncthreads();
  }

  // epilogue: bf16 store + per-channel sum / sumsq (BN1 batch statistics)
  const int rsub = (lane >> 4) << 3;
  const int cn   = lane & 15;
#pragma unroll
  for (int n = 0; n < 4; n++) {
    int col = colBase + wn * 64 + n * 16 + cn;
    float cs = 0.f, cq = 0.f;
#pragma unroll
    for (int m = 0; m < 2; m++) {
      int row0 = rowBase + wm * 32 + m * 16 + rsub;
#pragma unroll
      for (int r = 0; r < 8; r++) {
        float v = acc[m][n][r];
        h1[(size_t)(row0 + r) * EH + col] = (__bf16)v;
        cs += v; cq += v * v;
      }
    }
    cs += __shfl_xor(cs, 16, 32);         // lanes l / l+16 share a column
    cq += __shfl_xor(cq, 16, 32);
    if (lane < 16) { atomicAdd(&sum1[col], cs); atomicAdd(&sq1[col], cq); }
  }
}

// ---------------------------------------------------------------------------
__global__ void k_finalize(const float* __restrict__ sum, const float* __restrict__ sq,
                           const float* __restrict__ g, const float* __restrict__ b,
                           float* __restrict__ scale, float* __restrict__ shift, int n) {
  int i = blockIdx.x * blockDim.x + threadIdx.x;
  if (i < n) {
    float m   = sum[i] * (1.0f / B_N);
    float var = sq[i] * (1.0f / B_N) - m * m;
    float s   = rsqrtf(var + 1e-5f) * g[i];
    scale[i] = s;
    shift[i] = b[i] - m * s;
  }
}

// ---------------------------------------------------------------------------
__global__ __launch_bounds__(256) void k_gemm2(
    const __bf16* __restrict__ h1, const __bf16* __restrict__ w2t,
    const float* __restrict__ scale1, const float* __restrict__ shift1,
    __bf16* __restrict__ h2, float* __restrict__ sum2, float* __restrict__ sq2) {
  __shared__ __attribute__((aligned(16))) __bf16 sA[BM * LDT];
  __shared__ __attribute__((aligned(16))) __bf16 sB[BN * LDT];
  const int rowBase = blockIdx.x * BM;
  const int nTile   = blockIdx.y;           // 0..7 : 2 tiles per expert
  const int e       = nTile >> 1;
  const int colInE  = (nTile & 1) * BN;
  const int lane = threadIdx.x & 31;
  const int wave = threadIdx.x >> 5;
  const int wm = wave >> 1;
  const int wn = wave & 1;
  v8f acc[2][4] = {};

  for (int kk = 0; kk < H_DIM; kk += BK) {
#pragma unroll
    for (int i = 0; i < 2; i++) {
      int c = threadIdx.x + 256 * i;
      int r = c >> 2;
      int k8 = (c & 3) << 3;
      int kg = kk + k8;                     // k within this expert's H
      // B tile: pure copy -> async direct-to-LDS
      unsigned lb = lds_off(&sB[r * LDT + k8]);
      unsigned gb = (unsigned)((((size_t)e * O_DIM + colInE + r) * H_DIM + kg) * 2);
      async_copy_b128(lb, gb, w2t);
      // A tile: fused BN1 (scale/shift) + SiLU on the fly, bf16 into LDS
      v8bf raw = *(const v8bf*)&h1[(size_t)(rowBase + r) * EH + e * H_DIM + kg];
      if (kk + BK < H_DIM)                  // prefetch next A chunk
        __builtin_prefetch(&h1[(size_t)(rowBase + r) * EH + e * H_DIM + kg + BK], 0, 1);
      v8bf act;
#pragma unroll
      for (int j = 0; j < 8; j++) {
        int c1 = e * H_DIM + kg + j;
        float v = (float)raw[j];
        v = v * scale1[c1] + shift1[c1];
        v = v / (1.0f + __expf(-v));        // silu
        act[j] = (__bf16)v;
      }
      *(v8bf*)&sA[r * LDT + k8] = act;
    }
    wait_async0();
    __syncthreads();
    v16bf af[2], bf[4];
#pragma unroll
    for (int m = 0; m < 2; m++) af[m] = load_frag(&sA[(wm * 32 + m * 16) * LDT], lane);
#pragma unroll
    for (int n = 0; n < 4; n++) bf[n] = load_frag(&sB[(wn * 64 + n * 16) * LDT], lane);
#pragma unroll
    for (int m = 0; m < 2; m++)
#pragma unroll
      for (int n = 0; n < 4; n++)
        acc[m][n] = __builtin_amdgcn_wmma_f32_16x16x32_bf16(
            false, af[m], false, bf[n], (short)0, acc[m][n], false, false);
    __syncthreads();
  }

  const int rsub = (lane >> 4) << 3;
  const int cn   = lane & 15;
#pragma unroll
  for (int n = 0; n < 4; n++) {
    int colg = e * O_DIM + colInE + wn * 64 + n * 16 + cn;   // channel in [0,EO)
    float cs = 0.f, cq = 0.f;
#pragma unroll
    for (int m = 0; m < 2; m++) {
      int row0 = rowBase + wm * 32 + m * 16 + rsub;
#pragma unroll
      for (int r = 0; r < 8; r++) {
        float v = acc[m][n][r];
        h2[(size_t)(row0 + r) * EO + colg] = (__bf16)v;
        cs += v; cq += v * v;
      }
    }
    cs += __shfl_xor(cs, 16, 32);
    cq += __shfl_xor(cq, 16, 32);
    if (lane < 16) { atomicAdd(&sum2[colg], cs); atomicAdd(&sq2[colg], cq); }
  }
}

// ---------------------------------------------------------------------------
__device__ __forceinline__ float wave_sum(float v) {
#pragma unroll
  for (int off = 16; off > 0; off >>= 1) v += __shfl_xor(v, off, 32);
  return v;
}

__global__ __launch_bounds__(256) void k_epilogue(
    const float* __restrict__ x, const int* __restrict__ dom_ids,
    const __bf16* __restrict__ h2,
    const float* __restrict__ scale2, const float* __restrict__ shift2,
    const float* __restrict__ gW, const float* __restrict__ gb,
    const __bf16* __restrict__ tw1t, const float* __restrict__ tb1,
    const float* __restrict__ tW2, const float* __restrict__ tb2,
    float* __restrict__ out) {
  __shared__ float s_int[8][O_DIM];        // one interest row per wave
  const int wave = threadIdx.x >> 5;
  const int lane = threadIdx.x & 31;
  const int b    = blockIdx.x * 8 + wave;  // one sample per wave
  const int dom  = dom_ids[b];

  // gate logits: 4 dot products of x[b] with gate_W[dom][:,e]
  float acc[E_N] = {0.f, 0.f, 0.f, 0.f};
  const float* gw = gW + (size_t)dom * EMB * E_N;
  const float* xr = x + (size_t)b * EMB;
  for (int k = lane; k < EMB; k += 32) {
    float xv = xr[k];
#pragma unroll
    for (int e = 0; e < E_N; e++) acc[e] += xv * gw[k * E_N + e];
  }
#pragma unroll
  for (int e = 0; e < E_N; e++) acc[e] = wave_sum(acc[e]) + gb[dom * E_N + e];
  // softmax over experts (replicated across lanes)
  float mx = fmaxf(fmaxf(acc[0], acc[1]), fmaxf(acc[2], acc[3]));
  float g[E_N], gs = 0.f;
#pragma unroll
  for (int e = 0; e < E_N; e++) { g[e] = __expf(acc[e] - mx); gs += g[e]; }
  float inv = 1.0f / gs;
#pragma unroll
  for (int e = 0; e < E_N; e++) g[e] *= inv;

  // interest[o] = sum_e gate[e] * silu(bn2(h2[b, e, o]))
  for (int o = lane; o < O_DIM; o += 32) {
    float s = 0.f;
#pragma unroll
    for (int e = 0; e < E_N; e++) {
      int c = e * O_DIM + o;
      float v = (float)h2[(size_t)b * EO + c];
      v = v * scale2[c] + shift2[c];
      v = v / (1.0f + __expf(-v));
      s += g[e] * v;
    }
    s_int[wave][o] = s;
  }
  __syncthreads();

  // tower: relu(interest @ W1[dom] + b1) @ W2[dom] + b2
  float po = 0.f;
  const __bf16* tw = tw1t + (size_t)dom * TH_N * O_DIM;
  for (int th = lane; th < TH_N; th += 32) {
    float t = tb1[dom * TH_N + th];
    for (int o = 0; o < O_DIM; o++) t += s_int[wave][o] * (float)tw[th * O_DIM + o];
    t = fmaxf(t, 0.f);
    po += t * tW2[dom * TH_N + th];
  }
  po = wave_sum(po);
  if (lane == 0) out[b] = po + tb2[dom];
}

// ---------------------------------------------------------------------------
extern "C" void kernel_launch(void* const* d_in, const int* in_sizes, int n_in,
                              void* d_out, int out_size, void* d_ws, size_t ws_size,
                              hipStream_t stream) {
  (void)in_sizes; (void)n_in; (void)out_size; (void)ws_size;
  const float* x    = (const float*)d_in[0];
  const int*   dom  = (const int*)  d_in[1];
  const float* eW1  = (const float*)d_in[2];
  // d_in[3]=expert_b1, d_in[7]=expert_b2: cancel exactly under training-mode BN
  const float* bn1g = (const float*)d_in[4];
  const float* bn1b = (const float*)d_in[5];
  const float* eW2  = (const float*)d_in[6];
  const float* bn2g = (const float*)d_in[8];
  const float* bn2b = (const float*)d_in[9];
  const float* gW   = (const float*)d_in[10];
  const float* gb   = (const float*)d_in[11];
  const float* tW1  = (const float*)d_in[12];
  const float* tb1  = (const float*)d_in[13];
  const float* tW2  = (const float*)d_in[14];
  const float* tb2  = (const float*)d_in[15];
  float* out = (float*)d_out;

  char* ws = (char*)d_ws;
  size_t off = 0;
  __bf16* xb   = (__bf16*)(ws + off); off += (size_t)B_N * EMB * 2;          // 64 MiB
  __bf16* h1   = (__bf16*)(ws + off); off += (size_t)B_N * EH  * 2;          // 128 MiB
  __bf16* h2   = (__bf16*)(ws + off); off += (size_t)B_N * EO  * 2;          // 128 MiB
  __bf16* w1t  = (__bf16*)(ws + off); off += (size_t)EH * EMB * 2;           // 1 MiB
  __bf16* w2t  = (__bf16*)(ws + off); off += (size_t)E_N * O_DIM * H_DIM * 2;
  __bf16* tw1t = (__bf16*)(ws + off); off += (size_t)D_N * TH_N * O_DIM * 2;
  float* stats = (float*)(ws + off);                                         // 8 * 1024 f32
  float* sum1 = stats, *sq1 = stats + 1024, *scale1 = stats + 2048, *shift1 = stats + 3072;
  float* sum2 = stats + 4096, *sq2 = stats + 5120, *scale2 = stats + 6144, *shift2 = stats + 7168;

  k_prep<<<512, 256, 0, stream>>>(eW1, eW2, tW1, w1t, w2t, tw1t, stats);
  k_cvt_x<<<2048, 256, 0, stream>>>(x, xb);
  k_gemm1<<<dim3(B_N / BM, EH / BN), 256, 0, stream>>>(xb, w1t, h1, sum1, sq1);
  k_finalize<<<4, 256, 0, stream>>>(sum1, sq1, bn1g, bn1b, scale1, shift1, EH);
  k_gemm2<<<dim3(B_N / BM, (E_N * O_DIM) / BN), 256, 0, stream>>>(h1, w2t, scale1, shift1, h2, sum2, sq2);
  k_finalize<<<4, 256, 0, stream>>>(sum2, sq2, bn2g, bn2b, scale2, shift2, EO);
  k_epilogue<<<B_N / 8, 256, 0, stream>>>(x, dom, h2, scale2, shift2, gW, gb,
                                          tw1t, tb1, tW2, tb2, out);
}
